// GAT_53145925321198
// MI455X (gfx1250) — matrix operand order repaired and checked by reading, not compiled, not run on hardware
//
#include <hip/hip_runtime.h>
#include <hip/hip_bf16.h>
#include <math.h>
#include <stdint.h>

// ---------- types for WMMA ----------
typedef __attribute__((ext_vector_type(16))) _Float16 v16h;
typedef __attribute__((ext_vector_type(8)))  float    v8f;

#define NEG_SLOPE 0.2f

__device__ __forceinline__ float lrelu(float x) { return x > 0.f ? x : NEG_SLOPE * x; }

// float atomic-max via signed/unsigned int atomics (works with -inf init)
__device__ __forceinline__ void atomicMaxFloat(float* addr, float value) {
    if (value >= 0.f)
        atomicMax((int*)addr, __float_as_int(value));
    else
        atomicMin((unsigned int*)addr, __float_as_uint(value));
}

// ---------------------------------------------------------------------------
// Pre-pack W1 (128x64) and W2 (64x32) into f16 B-fragments, contiguous-k layout (this code's order):
// lane = 16*g + n  holds, in half slot (2v+j), element B[K][Ncol] with
//   K = kcBase + 16*g + 2*v + j ,  Ncol = tile*16 + n
// ---------------------------------------------------------------------------
__global__ void gat_prep_frags(const float* __restrict__ W1, const float* __restrict__ W2,
                               v16h* __restrict__ f1, v16h* __restrict__ f2) {
    int tid = blockIdx.x * blockDim.x + threadIdx.x;
    // W1: kc in 0..3, t in 0..3, lane in 0..31  -> 512 fragments
    if (tid < 4 * 4 * 32) {
        int lane = tid & 31, t = (tid >> 5) & 3, kc = tid >> 7;
        int g = lane >> 4, n = lane & 15;
        v16h b;
        #pragma unroll
        for (int v = 0; v < 8; ++v)
            #pragma unroll
            for (int j = 0; j < 2; ++j) {
                int k = kc * 32 + g * 16 + 2 * v + j;      // K index (0..127)
                b[2 * v + j] = (_Float16)W1[k * 64 + t * 16 + n];
            }
        f1[tid] = b;
    }
    // W2: kc in 0..1, t in 0..1, lane in 0..31 -> 128 fragments
    if (tid < 2 * 2 * 32) {
        int lane = tid & 31, t = (tid >> 5) & 1, kc = tid >> 6;
        int g = lane >> 4, n = lane & 15;
        v16h b;
        #pragma unroll
        for (int v = 0; v < 8; ++v)
            #pragma unroll
            for (int j = 0; j < 2; ++j) {
                int k = kc * 32 + g * 16 + 2 * v + j;      // K index (0..63)
                b[2 * v + j] = (_Float16)W2[k * 32 + t * 16 + n];
            }
        f2[tid] = b;
    }
}

// ---------------------------------------------------------------------------
// WMMA GEMM: C[nrows x OUT] = A[nrows x IN] * B   (B given as packed fragments)
// One wave (32 threads) per 16-row tile. A staged into LDS via CDNA5 async
// global->LDS copies (ASYNCcnt path), then converted to f16 fragments.
// ---------------------------------------------------------------------------
template <int IN, int OUT>
__global__ void gat_gemm_wmma(const float* __restrict__ A, const v16h* __restrict__ Bfrag,
                              float* __restrict__ C, int nrows) {
    constexpr int NT = OUT / 16;   // column tiles
    constexpr int KC = IN / 32;    // k-chunks of 32
    __shared__ float tl[16 * IN];

    const int lane = threadIdx.x & 31;
    const int row0 = blockIdx.x * 16;
    const int g = lane >> 4, n = lane & 15;

    constexpr int NF4 = 16 * IN / 4;           // 128b packets in one tile
    if (row0 + 16 <= nrows) {
        // fast path: async global -> LDS B128 copies, tracked by ASYNCcnt
        uint32_t lds_base = (uint32_t)(uintptr_t)tl;           // low 32b = LDS offset
        uint64_t gbase = (uint64_t)(uintptr_t)(A + (size_t)row0 * IN);
        #pragma unroll 1
        for (int i = lane; i < NF4; i += 32) {
            uint32_t loff = lds_base + (uint32_t)i * 16u;
            uint64_t gaddr = gbase + (uint64_t)i * 16u;
            asm volatile("global_load_async_to_lds_b128 %0, %1, off"
                         :: "v"(loff), "v"(gaddr) : "memory");
        }
        asm volatile("s_wait_asynccnt 0" ::: "memory");
    } else {
        // tail tile: guarded scalar staging
        const float4* src4 = (const float4*)(A + (size_t)row0 * IN);
        float4* dst4 = (float4*)tl;
        for (int i = lane; i < NF4; i += 32) {
            int r = i / (IN / 4);
            dst4[i] = (row0 + r < nrows) ? src4[i] : make_float4(0.f, 0.f, 0.f, 0.f);
        }
    }
    __syncthreads();   // single-wave workgroup: cheap cross-lane LDS fence

    v8f acc[NT];
    v8f zero = {};
    #pragma unroll
    for (int t = 0; t < NT; ++t) acc[t] = zero;

    #pragma unroll
    for (int kc = 0; kc < KC; ++kc) {
        // A fragment: lane (g, m=n) ; VGPR v -> K = kc*32 + 16*g + 2*v
        v16h a;
        #pragma unroll
        for (int v = 0; v < 8; ++v) {
            int K = kc * 32 + (g << 4) + (v << 1);
            float2 f = *(const float2*)(tl + n * IN + K);
            a[2 * v]     = (_Float16)f.x;
            a[2 * v + 1] = (_Float16)f.y;
        }
        #pragma unroll
        for (int t = 0; t < NT; ++t) {
            v16h b = Bfrag[(size_t)(kc * NT + t) * 32 + lane];
            acc[t] = __builtin_amdgcn_wmma_f32_16x16x32_f16(
                false, a, false, b, (short)0, acc[t], false, false);
        }
    }

    // store: VGPR r of acc -> row M = r + 8*g, col = t*16 + n
    #pragma unroll
    for (int t = 0; t < NT; ++t)
        #pragma unroll
        for (int r = 0; r < 8; ++r) {
            int row = row0 + r + 8 * g;
            if (row < nrows) C[(size_t)row * OUT + t * 16 + n] = acc[t][r];
        }
}

// ---------------------------------------------------------------------------
// Per-node prep: el/er dot products, init max=-inf, sum=0, zero aggregation buf
// hmat layout: [n][H][32]
// ---------------------------------------------------------------------------
__global__ void gat_node_prep(const float* __restrict__ hmat,
                              const float* __restrict__ al, const float* __restrict__ ar,
                              float* __restrict__ el, float* __restrict__ er,
                              float* __restrict__ mx, float* __restrict__ sm,
                              float* __restrict__ outz, int nnodes, int H) {
    int nn = blockIdx.x * blockDim.x + threadIdx.x;
    if (nn >= nnodes) return;
    for (int h = 0; h < H; ++h) {
        const float* hr = hmat + ((size_t)nn * H + h) * 32;
        float accl = 0.f, accr = 0.f;
        #pragma unroll
        for (int d = 0; d < 32; ++d) {
            float hv = hr[d];
            accl += hv * al[h * 32 + d];
            accr += hv * ar[h * 32 + d];
        }
        el[nn * H + h] = accl;
        er[nn * H + h] = accr;
        mx[nn * H + h] = -INFINITY;
        sm[nn * H + h] = 0.f;
    }
    for (int j = 0; j < H * 32; ++j) outz[(size_t)nn * H * 32 + j] = 0.f;
}

// ---------------------------------------------------------------------------
// Edge passes
// ---------------------------------------------------------------------------
__global__ void gat_edge_max(const long long* __restrict__ src, const long long* __restrict__ dst,
                             const float* __restrict__ el, const float* __restrict__ er,
                             float* __restrict__ mx, int E, int H) {
    int i = blockIdx.x * blockDim.x + threadIdx.x;
    if (i >= E) return;
    long long s = src[i], d = dst[i];
    for (int h = 0; h < H; ++h) {
        float e = lrelu(el[s * H + h] + er[d * H + h]);
        atomicMaxFloat(&mx[d * H + h], e);
    }
}

__global__ void gat_edge_sum(const long long* __restrict__ src, const long long* __restrict__ dst,
                             const float* __restrict__ el, const float* __restrict__ er,
                             const float* __restrict__ mx, float* __restrict__ sm, int E, int H) {
    int i = blockIdx.x * blockDim.x + threadIdx.x;
    if (i >= E) return;
    long long s = src[i], d = dst[i];
    for (int h = 0; h < H; ++h) {
        float e = lrelu(el[s * H + h] + er[d * H + h]);
        atomicAdd(&sm[d * H + h], __expf(e - mx[d * H + h]));
    }
}

// lane-per-dim message scatter: a wave covers one edge's contiguous features,
// so h[src] gathers are full 128B lines and atomics coalesce per cacheline.
template <int H>
__global__ void gat_edge_msg(const long long* __restrict__ src, const long long* __restrict__ dst,
                             const float* __restrict__ el, const float* __restrict__ er,
                             const float* __restrict__ mx, const float* __restrict__ sm,
                             const float* __restrict__ hin, float* __restrict__ out,
                             long long total) {
    constexpr int C = H * 32;
    long long tid = (long long)blockIdx.x * blockDim.x + threadIdx.x;
    if (tid >= total) return;
    long long e = tid / C;               // C is 32 or 64 -> shift
    int j = (int)(tid - e * C);          // feature index within node row
    int h = j >> 5;                      // head
    long long s = src[e], d = dst[e];    // same-address across wave -> broadcast
    float att = lrelu(el[s * H + h] + er[d * H + h]);
    float a = __expf(att - mx[d * H + h]) / sm[d * H + h];
    atomicAdd(out + (size_t)d * C + j, hin[(size_t)s * C + j] * a);
}

// out[i] = relu(in[i] + b[i & mask])   (mask = C-1, C power of two)
__global__ void gat_bias_relu(const float* __restrict__ in, const float* __restrict__ b,
                              float* __restrict__ out, int total, int mask) {
    int i = blockIdx.x * blockDim.x + threadIdx.x;
    if (i >= total) return;
    float v = in[i] + b[i & mask];
    out[i] = v > 0.f ? v : 0.f;
}

// ---------------------------------------------------------------------------
extern "C" void kernel_launch(void* const* d_in, const int* in_sizes, int n_in,
                              void* d_out, int out_size, void* d_ws, size_t ws_size,
                              hipStream_t stream) {
    const float* feat = (const float*)d_in[0];
    const float* W1   = (const float*)d_in[1];
    const float* al1  = (const float*)d_in[2];
    const float* ar1  = (const float*)d_in[3];
    const float* b1   = (const float*)d_in[4];
    const float* W2   = (const float*)d_in[5];
    const float* al2  = (const float*)d_in[6];
    const float* ar2  = (const float*)d_in[7];
    const float* b2   = (const float*)d_in[8];
    const long long* src = (const long long*)d_in[9];
    const long long* dst = (const long long*)d_in[10];
    float* out = (float*)d_out;

    const int N = in_sizes[0] / 128;   // 100000 (divisible by 16)
    const int E = in_sizes[9];         // 1,700,000

    // workspace layout (floats): 204 per node + weight fragments
    float* ws  = (float*)d_ws;
    float* h1  = ws;                           // N*64
    float* o1  = h1 + (size_t)N * 64;          // N*64
    float* el1 = o1 + (size_t)N * 64;          // N*2
    float* er1 = el1 + (size_t)N * 2;
    float* m1  = er1 + (size_t)N * 2;
    float* s1  = m1  + (size_t)N * 2;
    float* h2  = s1  + (size_t)N * 2;          // N*32
    float* o2  = h2  + (size_t)N * 32;         // N*32
    float* el2 = o2  + (size_t)N * 32;         // N
    float* er2 = el2 + (size_t)N;
    float* m2  = er2 + (size_t)N;
    float* s2  = m2  + (size_t)N;
    uintptr_t p = (uintptr_t)(s2 + (size_t)N);
    p = (p + 31) & ~(uintptr_t)31;
    v16h* f1 = (v16h*)p;                        // 4*4*32 fragments
    v16h* f2 = f1 + 4 * 4 * 32;                 // 2*2*32 fragments

    const int ntile = (N + 15) / 16;

    // weight fragment pack
    gat_prep_frags<<<1, 512, 0, stream>>>(W1, W2, f1, f2);

    // ---- layer 1 ----
    gat_gemm_wmma<128, 64><<<ntile, 32, 0, stream>>>(feat, f1, h1, N);
    gat_node_prep<<<(N + 255) / 256, 256, 0, stream>>>(h1, al1, ar1, el1, er1, m1, s1, o1, N, 2);
    gat_edge_max<<<(E + 255) / 256, 256, 0, stream>>>(src, dst, el1, er1, m1, E, 2);
    gat_edge_sum<<<(E + 255) / 256, 256, 0, stream>>>(src, dst, el1, er1, m1, s1, E, 2);
    {
        long long total = (long long)E * 64;
        int blocks = (int)((total + 255) / 256);
        gat_edge_msg<2><<<blocks, 256, 0, stream>>>(src, dst, el1, er1, m1, s1, h1, o1, total);
    }
    gat_bias_relu<<<(N * 64 + 255) / 256, 256, 0, stream>>>(o1, b1, o1, N * 64, 63);

    // ---- layer 2 ----
    gat_gemm_wmma<64, 32><<<ntile, 32, 0, stream>>>(o1, f2, h2, N);
    gat_node_prep<<<(N + 255) / 256, 256, 0, stream>>>(h2, al2, ar2, el2, er2, m2, s2, o2, N, 1);
    gat_edge_max<<<(E + 255) / 256, 256, 0, stream>>>(src, dst, el2, er2, m2, E, 1);
    gat_edge_sum<<<(E + 255) / 256, 256, 0, stream>>>(src, dst, el2, er2, m2, s2, E, 1);
    {
        long long total = (long long)E * 32;
        int blocks = (int)((total + 255) / 256);
        gat_edge_msg<1><<<blocks, 256, 0, stream>>>(src, dst, el2, er2, m2, s2, h2, o2, total);
    }
    gat_bias_relu<<<(N * 32 + 255) / 256, 256, 0, stream>>>(o2, b2, out, N * 32, 31);
}
